// CrossDynamicConv_77051713290514
// MI455X (gfx1250) — matrix operand order repaired and checked by reading, not compiled, not run on hardware
//
#include <hip/hip_runtime.h>
#include <hip/hip_bf16.h>

// Sizes from the reference
#define NN   1024
#define CC   256
#define FF   64
#define HWW  49
#define OUTC 256
#define KDYN 256
#define NDYN 32768   // C*F + F*OUT

typedef __bf16 bf16;
typedef __attribute__((ext_vector_type(16))) __bf16 v16bf;
typedef __attribute__((ext_vector_type(8)))  __bf16 v8bf;
typedef __attribute__((ext_vector_type(8)))  float  v8f;
typedef __attribute__((ext_vector_type(2)))  float  v2f;

__device__ __forceinline__ bf16 f2bf(float x) { return (bf16)x; }

// CDNA5 async global->LDS copy (ASYNCcnt-tracked), ISA 08_async_tensor §4.
// vdst = 32-bit LDS byte offset, vaddr = 64-bit global address (GV mode).
__device__ __forceinline__ void async_g2l_b128(void* lds, const void* g) {
  unsigned l = (unsigned)(size_t)lds;   // low 32 bits of generic ptr = LDS offset
  asm volatile("global_load_async_to_lds_b128 %0, %1, off"
               :: "v"(l), "v"(g) : "memory");
}
__device__ __forceinline__ void wait_async3() {
  asm volatile("s_wait_asynccnt 0x3" ::: "memory");
}
__device__ __forceinline__ void wait_async0() {
  asm volatile("s_wait_asynccnt 0x0" ::: "memory");
}

// A fragment (16-bit A 16x32, ISA 7.12.2): lanes 0-15 hold K{0..7,16..23},
// lanes 16-31 hold K{8..15,24..31}. p = rowbase + k0 + half*8.
__device__ __forceinline__ v16bf ldfragA(const bf16* p) {
  union { v16bf v; v8bf h[2]; } u;
  u.h[0] = *(const v8bf*)(p);
  u.h[1] = *(const v8bf*)(p + 16);
  return u.v;
}
// B fragment (16-bit B 32x16): lanes 0-15 hold K{0..7,16..23},
// lanes 16-31 hold K{8..15,24..31}. p = colbase + k0 + half*8.
__device__ __forceinline__ v16bf ldfragB(const bf16* p) {
  union { v16bf v; v8bf h[2]; } u;
  u.h[0] = *(const v8bf*)(p);
  u.h[1] = *(const v8bf*)(p + 16);
  return u.v;
}

// ---------------------------------------------------------------------------
// Kernel 1: params = pf[1024,256] @ W_dyn[256,32768] + b_dyn  (exact f32 WMMA)
// writes directly into output p_in [N,C,F] / p_out [N,F,OUT] regions.
// Memory-bound (32MB weight read + 128MB write): stage tiles with
// global_load_async_to_lds_b128, double-buffered, overlapped with WMMA.
// Block: 32x64 tile, 8 waves (2x4), K-step 32 (8x v_wmma_f32_16x16x4_f32).
// ---------------------------------------------------------------------------
__global__ __launch_bounds__(256) void k_dyn(const float* __restrict__ A,
    const float* __restrict__ W, const float* __restrict__ bias,
    float* __restrict__ pin, float* __restrict__ pout) {
  __shared__ __align__(16) float sA[2][32 * 36];   // padded row-major [m][k]
  __shared__ __align__(16) float sB[2][32 * 64];   // packed row-major [k][n]
  const int t = threadIdx.x;
  const int w = t >> 5, lane = t & 31, half = lane >> 4, ln = lane & 15;
  const int wm = w >> 2, wn = w & 3;
  const int n0 = blockIdx.x * 64, m0 = blockIdx.y * 32;

  // per-thread async-staging coordinates (3 b128 issues / thread / stage)
  const int ar = t >> 3, ak = (t & 7) * 4;           // A: 32x32 tile
  const int bk0 = t >> 4, bnq = (t & 15) * 4;        // B: chunks t and t+256

  auto issue = [&](int s, int k0) {
    async_g2l_b128(&sA[s][ar*36 + ak], A + (size_t)(m0 + ar) * KDYN + k0 + ak);
    async_g2l_b128(&sB[s][bk0*64 + bnq],
                   W + (size_t)(k0 + bk0) * NDYN + n0 + bnq);
    async_g2l_b128(&sB[s][(bk0 + 16)*64 + bnq],
                   W + (size_t)(k0 + bk0 + 16) * NDYN + n0 + bnq);
  };

  v8f acc = {};
  issue(0, 0);
  int s = 0;
  for (int k0 = 0; k0 < KDYN; k0 += 32, s ^= 1) {
    if (k0 + 32 < KDYN) { issue(s ^ 1, k0 + 32); wait_async3(); }
    else                { wait_async0(); }
    __syncthreads();
    #pragma unroll
    for (int kk = 0; kk < 32; kk += 4) {
      v2f a, b;
      a[0] = sA[s][(wm*16+ln)*36 + kk + half*2];
      a[1] = sA[s][(wm*16+ln)*36 + kk + half*2 + 1];
      b[0] = sB[s][(kk + half*2)*64     + wn*16 + ln];
      b[1] = sB[s][(kk + half*2 + 1)*64 + wn*16 + ln];
      acc = __builtin_amdgcn_wmma_f32_16x16x4_f32(false, a, false, b,
                                                  (short)0, acc, false, false);
    }
    __syncthreads();
  }
  // block-uniform output select: the p_in/p_out split (16384) is a multiple
  // of the 64-wide tile, and both have row stride 16384 -> branch-free stores.
  float* base = (n0 < CC * FF) ? (pin + n0) : (pout + (n0 - CC * FF));
  const int col = wn*16 + ln;
  const float bv = bias[n0 + col];
  #pragma unroll
  for (int v = 0; v < 8; ++v) {
    int row = m0 + wm*16 + half*8 + v;
    base[(size_t)row * (CC * FF) + col] = acc[v] + bv;
  }
}

// ---------------------------------------------------------------------------
// Kernel 2: sigmoid gates at l = L-1 = 3 only (live slice after DCE).
// out[n,c] = sigmoid(pf[n,:] @ W[:, c*4+3] + b[c*4+3]).  f32 WMMA.
// ---------------------------------------------------------------------------
__global__ __launch_bounds__(256) void k_gate(const float* __restrict__ A,
    const float* __restrict__ W, const float* __restrict__ bias,
    float* __restrict__ out, int Ncols) {
  __shared__ float sA[32 * 36];
  __shared__ float sB[64 * 36];
  const int t = threadIdx.x;
  const int w = t >> 5, lane = t & 31, half = lane >> 4, ln = lane & 15;
  const int wm = w >> 2, wn = w & 3;
  const int n0 = blockIdx.x * 64, m0 = blockIdx.y * 32;
  const int ldW = Ncols * 4;
  v8f acc = {};
  for (int k0 = 0; k0 < KDYN; k0 += 32) {
    { int r = t >> 3, k = (t & 7) * 4;
      const float* p = A + (size_t)(m0 + r) * KDYN + k0 + k;
      sA[r*36+k+0] = p[0]; sA[r*36+k+1] = p[1];
      sA[r*36+k+2] = p[2]; sA[r*36+k+3] = p[3]; }
    { int nn = t & 63, kk = t >> 6;
      #pragma unroll
      for (int j = 0; j < 8; ++j) {
        int k = kk * 8 + j;
        sB[nn*36 + k] = W[(size_t)(k0 + k) * ldW + (n0 + nn)*4 + 3];
      } }
    __syncthreads();
    #pragma unroll
    for (int kk = 0; kk < 32; kk += 4) {
      v2f a, b;
      a[0] = sA[(wm*16+ln)*36 + kk + half*2];
      a[1] = sA[(wm*16+ln)*36 + kk + half*2 + 1];
      b[0] = sB[(wn*16+ln)*36 + kk + half*2];
      b[1] = sB[(wn*16+ln)*36 + kk + half*2 + 1];
      acc = __builtin_amdgcn_wmma_f32_16x16x4_f32(false, a, false, b,
                                                  (short)0, acc, false, false);
    }
    __syncthreads();
  }
  const int col = n0 + wn*16 + ln;
  const float bv = bias[col*4 + 3];
  #pragma unroll
  for (int v = 0; v < 8; ++v) {
    int row = m0 + wm*16 + half*8 + v;
    float val = acc[v] + bv;
    out[(size_t)row * Ncols + col] = 1.0f / (1.0f + __expf(-val));
  }
}

// ---------------------------------------------------------------------------
// Kernel 3: fused per-sample interaction (block = proposal n, 8 waves).
//   u_in  = p_in  + m1*m2*(pre_in[3]  - p_in)   (collapsed two-step gate mix)
//   u_out = p_out + s1*s2*(pre_out[3] - p_out)
//   f = relu(ln(x @ u_in)); f = relu(ln(f @ u_out)) -> gf[N,49,256]
// bf16 WMMA 16x16x32, f32 accum; LN is scale-invariant so bf16 inputs are fine.
// ---------------------------------------------------------------------------
__global__ __launch_bounds__(256) void k_inter(
    const float* __restrict__ xin,      // input_feature [N,C,49]
    const float* __restrict__ pin, const float* __restrict__ pout,
    const float* __restrict__ pre_in, const float* __restrict__ pre_out,
    const float* __restrict__ gm1, const float* __restrict__ gm2,
    const float* __restrict__ gs1, const float* __restrict__ gs2,
    const float* __restrict__ g_in, const float* __restrict__ b_in,
    const float* __restrict__ g_out, const float* __restrict__ b_out,
    float* __restrict__ gf) {
  __shared__ bf16  sA [64 * 264];   // x_n   [h][c]  row-major, padded
  __shared__ bf16  sB [64 * 264];   // u_in  [f][c]  (B stored per-column)
  __shared__ float sF1[64 * 66];    // f1    [h][f]
  __shared__ bf16  sA2[64 * 72];    // ln(f1)[h][f]
  __shared__ bf16  sB2[256 * 72];   // u_out [o][f]
  __shared__ float sF2[64 * 260];   // f2    [h][o]
  __shared__ float sm1[64], ss1[64];

  const int n = blockIdx.x;
  const int t = threadIdx.x;
  const int w = t >> 5, lane = t & 31, half = lane >> 4, ln = lane & 15;

  if (t < 64)       sm1[t]      = gm1[n*64 + t];
  else if (t < 128) ss1[t - 64] = gs1[n*64 + (t - 64)];
  __syncthreads();

  // stage x_n transposed: sA[h][c] = input_feature[n][c][h]
  { const float* xp = xin + ((size_t)n * CC + t) * HWW;
    for (int h = 0; h < HWW; ++h) sA[h*264 + t] = f2bf(xp[h]);
    for (int h = HWW; h < 64; ++h) sA[h*264 + t] = f2bf(0.0f); }
  // stage u_in as B tile: sB[f][c]
  { float m2v = gm2[(size_t)n * CC + t];
    const float* pp = pin    + ((size_t)n * CC + t) * FF;
    const float* qq = pre_in + ((size_t)(3*NN + n) * CC + t) * FF;
    for (int f = 0; f < FF; ++f) {
      float p = pp[f];
      sB[f*264 + t] = f2bf(p + sm1[f] * m2v * (qq[f] - p));
    } }
  // stage u_out as B tile: sB2[o][f]
  { float s2v = gs2[(size_t)n * OUTC + t];
    const float* pp = pout    + (size_t)n * FF * OUTC + t;
    const float* qq = pre_out + (size_t)(3*NN + n) * FF * OUTC + t;
    for (int f = 0; f < FF; ++f) {
      float p = pp[(size_t)f * OUTC];
      sB2[t*72 + f] = f2bf(p + ss1[f] * s2v * (qq[(size_t)f * OUTC] - p));
    } }
  // zero pad rows 49..63 of GEMM2's A operand
  for (int i = t; i < 15 * 72; i += 256) sA2[49*72 + i] = f2bf(0.0f);
  __syncthreads();

  // GEMM1: [64h x 64f], K=256.  16 wave-tiles over 8 waves.
  for (int id = w; id < 16; id += 8) {
    int tm = id >> 2, tn = id & 3;
    v8f acc = {};
    #pragma unroll
    for (int k0 = 0; k0 < 256; k0 += 32) {
      v16bf a = ldfragA(sA + (tm*16 + ln)*264 + k0 + half*8);
      v16bf b = ldfragB(sB + (tn*16 + ln)*264 + k0 + half*8);
      acc = __builtin_amdgcn_wmma_f32_16x16x32_bf16(false, a, false, b,
                                                    (short)0, acc, false, false);
    }
    #pragma unroll
    for (int v = 0; v < 8; ++v)
      sF1[(tm*16 + half*8 + v)*66 + tn*16 + ln] = acc[v];
  }
  __syncthreads();

  // LN over F=64 + ReLU (wave per row), emit bf16 A for GEMM2
  for (int r = w; r < HWW; r += 8) {
    float x0 = sF1[r*66 + lane];
    float x1 = sF1[r*66 + 32 + lane];
    float s = x0 + x1, q = x0*x0 + x1*x1;
    #pragma unroll
    for (int off = 16; off; off >>= 1) {
      s += __shfl_xor(s, off, 32); q += __shfl_xor(q, off, 32);
    }
    float mean = s * (1.0f/64.0f);
    float inv  = rsqrtf(q * (1.0f/64.0f) - mean*mean + 1e-5f);
    sA2[r*72 + lane]      = f2bf(fmaxf((x0-mean)*inv*g_in[lane]    + b_in[lane],    0.0f));
    sA2[r*72 + 32 + lane] = f2bf(fmaxf((x1-mean)*inv*g_in[32+lane] + b_in[32+lane], 0.0f));
  }
  __syncthreads();

  // GEMM2: [64h x 256o], K=64.  64 wave-tiles over 8 waves.
  for (int id = w; id < 64; id += 8) {
    int tm = id >> 4, tn = id & 15;
    v8f acc = {};
    #pragma unroll
    for (int k0 = 0; k0 < 64; k0 += 32) {
      v16bf a = ldfragA(sA2 + (tm*16 + ln)*72 + k0 + half*8);
      v16bf b = ldfragB(sB2 + (tn*16 + ln)*72 + k0 + half*8);
      acc = __builtin_amdgcn_wmma_f32_16x16x32_bf16(false, a, false, b,
                                                    (short)0, acc, false, false);
    }
    #pragma unroll
    for (int v = 0; v < 8; ++v)
      sF2[(tm*16 + half*8 + v)*260 + tn*16 + ln] = acc[v];
  }
  __syncthreads();

  // LN over OUT=256 + ReLU, write gf
  float* go = gf + (size_t)n * HWW * OUTC;
  for (int r = w; r < HWW; r += 8) {
    float xs[8]; float s = 0.0f, q = 0.0f;
    #pragma unroll
    for (int j = 0; j < 8; ++j) {
      float x = sF2[r*260 + j*32 + lane]; xs[j] = x; s += x; q += x*x;
    }
    #pragma unroll
    for (int off = 16; off; off >>= 1) {
      s += __shfl_xor(s, off, 32); q += __shfl_xor(q, off, 32);
    }
    float mean = s * (1.0f/256.0f);
    float inv  = rsqrtf(q * (1.0f/256.0f) - mean*mean + 1e-5f);
    #pragma unroll
    for (int j = 0; j < 8; ++j) {
      int c = j*32 + lane;
      go[(size_t)r * OUTC + c] = fmaxf((xs[j]-mean)*inv*g_out[c] + b_out[c], 0.0f);
    }
  }
}

// ---------------------------------------------------------------------------
// Kernel 4: out = relu(ln(gf[N,12544] @ W_fc[12544,256] + b_fc))
// Block: 16 rows x full 256 cols (so LN fuses in-epilogue). bf16 WMMA.
// ---------------------------------------------------------------------------
__global__ __launch_bounds__(256) void k_fc(
    const float* __restrict__ gf, const float* __restrict__ Wfc,
    const float* __restrict__ bfc, const float* __restrict__ gln,
    const float* __restrict__ bln, float* __restrict__ out) {
  __shared__ bf16  sAt[16 * 40];
  __shared__ bf16  sBt[256 * 40];
  __shared__ float sO [16 * 264];
  const int t = threadIdx.x;
  const int w = t >> 5, lane = t & 31, half = lane >> 4, ln = lane & 15;
  const int m0 = blockIdx.x * 16;
  v8f acc0 = {}, acc1 = {};
  for (int k0 = 0; k0 < HWW * OUTC; k0 += 32) {
    { int r = t >> 4, k = (t & 15) * 2;
      const float* p = gf + (size_t)(m0 + r) * (HWW*OUTC) + k0 + k;
      sAt[r*40 + k]     = f2bf(p[0]);
      sAt[r*40 + k + 1] = f2bf(p[1]); }
    #pragma unroll 4
    for (int k = 0; k < 32; ++k)
      sBt[t*40 + k] = f2bf(Wfc[(size_t)(k0 + k) * 256 + t]);
    __syncthreads();
    v16bf a  = ldfragA(sAt + ln*40 + half*8);
    v16bf b0 = ldfragB(sBt + (w*32 + ln)*40 + half*8);
    v16bf b1 = ldfragB(sBt + (w*32 + 16 + ln)*40 + half*8);
    acc0 = __builtin_amdgcn_wmma_f32_16x16x32_bf16(false, a, false, b0,
                                                   (short)0, acc0, false, false);
    acc1 = __builtin_amdgcn_wmma_f32_16x16x32_bf16(false, a, false, b1,
                                                   (short)0, acc1, false, false);
    __syncthreads();
  }
  #pragma unroll
  for (int v = 0; v < 8; ++v) {
    sO[(half*8 + v)*264 + w*32 + ln]      = acc0[v] + bfc[w*32 + ln];
    sO[(half*8 + v)*264 + w*32 + 16 + ln] = acc1[v] + bfc[w*32 + 16 + ln];
  }
  __syncthreads();
  for (int r = w; r < 16; r += 8) {
    float xs[8]; float s = 0.0f, q = 0.0f;
    #pragma unroll
    for (int j = 0; j < 8; ++j) {
      float x = sO[r*264 + j*32 + lane]; xs[j] = x; s += x; q += x*x;
    }
    #pragma unroll
    for (int off = 16; off; off >>= 1) {
      s += __shfl_xor(s, off, 32); q += __shfl_xor(q, off, 32);
    }
    float mean = s * (1.0f/256.0f);
    float inv  = rsqrtf(q * (1.0f/256.0f) - mean*mean + 1e-5f);
    #pragma unroll
    for (int j = 0; j < 8; ++j) {
      int c = j*32 + lane;
      out[(size_t)(m0 + r) * 256 + c] = fmaxf((xs[j]-mean)*inv*gln[c] + bln[c], 0.0f);
    }
  }
}

// ---------------------------------------------------------------------------
extern "C" void kernel_launch(void* const* d_in, const int* in_sizes, int n_in,
                              void* d_out, int out_size, void* d_ws, size_t ws_size,
                              hipStream_t stream) {
  (void)in_sizes; (void)n_in; (void)out_size; (void)ws_size;
  const float* pf    = (const float*)d_in[0];
  const float* xin   = (const float*)d_in[1];
  const float* prei  = (const float*)d_in[2];
  const float* preo  = (const float*)d_in[3];
  const float* Wdyn  = (const float*)d_in[4];
  const float* bdyn  = (const float*)d_in[5];
  const float* g_in  = (const float*)d_in[6];
  const float* b_in  = (const float*)d_in[7];
  const float* g_out = (const float*)d_in[8];
  const float* b_out = (const float*)d_in[9];
  const float* Wfc   = (const float*)d_in[10];
  const float* bfc   = (const float*)d_in[11];
  const float* gfcn  = (const float*)d_in[12];
  const float* bfcn  = (const float*)d_in[13];
  const float* WM1   = (const float*)d_in[14];
  const float* bM1   = (const float*)d_in[15];
  const float* WM2   = (const float*)d_in[16];
  const float* bM2   = (const float*)d_in[17];
  const float* WS1   = (const float*)d_in[18];
  const float* bS1   = (const float*)d_in[19];
  const float* WS2   = (const float*)d_in[20];
  const float* bS2   = (const float*)d_in[21];
  // d_in[22..25] (W_sh/b_sh/W_sw/b_sw) are dead code for the returned outputs.

  float* f_out = (float*)d_out;                       // [N, 256]
  float* pin   = f_out + (size_t)NN * OUTC;           // [N, C, F]
  float* pout  = pin   + (size_t)NN * CC * FF;        // [N, F, OUT]

  float* ws  = (float*)d_ws;
  float* gm1 = ws;                                    // [N, 64]
  float* gm2 = gm1 + (size_t)NN * FF;                 // [N, 256]
  float* gs1 = gm2 + (size_t)NN * CC;                 // [N, 64]
  float* gs2 = gs1 + (size_t)NN * FF;                 // [N, 256]
  float* gf  = gs2 + (size_t)NN * OUTC;               // [N, 49, 256]

  k_dyn <<<dim3(NDYN/64, NN/32), 256, 0, stream>>>(pf, Wdyn, bdyn, pin, pout);
  k_gate<<<dim3(1, NN/32), 256, 0, stream>>>(pf, WM1, bM1, gm1, 64);
  k_gate<<<dim3(4, NN/32), 256, 0, stream>>>(pf, WM2, bM2, gm2, 256);
  k_gate<<<dim3(1, NN/32), 256, 0, stream>>>(pf, WS1, bS1, gs1, 64);
  k_gate<<<dim3(4, NN/32), 256, 0, stream>>>(pf, WS2, bS2, gs2, 256);
  k_inter<<<NN, 256, 0, stream>>>(xin, pin, pout, prei, preo,
                                  gm1, gm2, gs1, gs2,
                                  g_in, b_in, g_out, b_out, gf);
  k_fc  <<<64, 256, 0, stream>>>(gf, Wfc, bfc, gfcn, bfcn, f_out);
}
